// AttentionBlock_4054449127949
// MI455X (gfx1250) — matrix operand order repaired and checked by reading, not compiled, not run on hardware
//
#include <hip/hip_runtime.h>
#include <hip/hip_bf16.h>
#include <math.h>

// ---------------------------------------------------------------------------
// AttentionBlock for MI455X (gfx1250, wave32).
// All GEMMs: v_wmma_f32_16x16x32_f16, 16x128 tile per wave (8 acc, A reuse x8).
// Inner loop: load all 8 B fragments (b128 clauses), then 8 back-to-back WMMAs.
// Two-pass attention: S (f16, 32MB/batch) stays resident in the 192MB L2.
// ---------------------------------------------------------------------------

typedef __attribute__((ext_vector_type(16))) _Float16 v16h;
typedef __attribute__((ext_vector_type(8)))  _Float16 h8;
typedef __attribute__((ext_vector_type(8)))  float    f8;
typedef __attribute__((ext_vector_type(8)))  float    v8f;

#define B_ 4
#define C_ 512
#define N_ 4096
#define G_ 32
#define NJ 8   // 16x(NJ*16) output tile per wave

#define SHUF16(lo, hi) __builtin_shufflevector((lo), (hi), 0, 1, 2, 3, 4, 5, 6, 7, \
                                               8, 9, 10, 11, 12, 13, 14, 15)

// ---- WMMA fragment loaders (ISA 7.12.2 layouts, wave32) --------------------
// A fragment 16x32 (MxK), element (m,k) at A[m*lda + k] (row-major, k contig).
// lane 0-15: M=lane, K={0..7,16..23}; lane 16-31: M=lane-16, K={8..15,24..31}
static __device__ __forceinline__ v16h frag_a_f16(const _Float16* __restrict__ A,
                                                  int lda, int m0, int k0, int lane) {
  const _Float16* p = A + (size_t)(m0 + (lane & 15)) * lda + k0 + ((lane >> 4) << 3);
  h8 lo = *(const h8*)p;          // K chunk +0..7
  h8 hi = *(const h8*)(p + 16);   // K chunk +16..23
  return SHUF16(lo, hi);
}

static __device__ __forceinline__ v16h frag_a_f32(const float* __restrict__ A,
                                                  int lda, int m0, int k0, int lane) {
  const float* p = A + (size_t)(m0 + (lane & 15)) * lda + k0 + ((lane >> 4) << 3);
  h8 lo = __builtin_convertvector(*(const f8*)p, h8);
  h8 hi = __builtin_convertvector(*(const f8*)(p + 16), h8);
  return SHUF16(lo, hi);
}

// B fragment 32x16 (KxN), element (k,n) stored n-major at Bt[n*ld + k]
// -> per lane: two 8-half chunks (two b128 loads).
// lane 0-15: N=lane, K={0..7,16..23}; lane 16-31: N=lane-16, K={8..15,24..31}
static __device__ __forceinline__ v16h frag_b_nmajor(const _Float16* __restrict__ Bt,
                                                     int ld, int k0, int n0, int lane) {
  const _Float16* p = Bt + (size_t)(n0 + (lane & 15)) * ld + k0 + ((lane >> 4) << 3);
  h8 lo = *(const h8*)p;
  h8 hi = *(const h8*)(p + 16);
  return SHUF16(lo, hi);
}

#define WMMA_F16(a, b, c) \
  __builtin_amdgcn_wmma_f32_16x16x32_f16(false, (a), false, (b), (short)0, (c), false, false)

#define ZACC8 {{0.f,0.f,0.f,0.f,0.f,0.f,0.f,0.f}, {0.f,0.f,0.f,0.f,0.f,0.f,0.f,0.f}, \
               {0.f,0.f,0.f,0.f,0.f,0.f,0.f,0.f}, {0.f,0.f,0.f,0.f,0.f,0.f,0.f,0.f}, \
               {0.f,0.f,0.f,0.f,0.f,0.f,0.f,0.f}, {0.f,0.f,0.f,0.f,0.f,0.f,0.f,0.f}, \
               {0.f,0.f,0.f,0.f,0.f,0.f,0.f,0.f}, {0.f,0.f,0.f,0.f,0.f,0.f,0.f,0.f}}

// ---------------------------------------------------------------------------
// Kernel 1: GroupNorm (32 groups). One block per (b,g).
// Input x f32 [B][C][N]; output hn f16 [B][N][C] (n-major for GEMM B-operand).
// ---------------------------------------------------------------------------
__global__ void gn_kernel(const float* __restrict__ x, const float* __restrict__ gamma,
                          const float* __restrict__ beta, _Float16* __restrict__ hn) {
  const int CPG = C_ / G_;          // 16
  const int M   = CPG * N_;         // 65536 elements per (b,g)
  int b = blockIdx.x / G_;
  int g = blockIdx.x % G_;
  size_t base = ((size_t)b * C_ + (size_t)g * CPG) * N_;
  int lane = threadIdx.x & 31, wid = threadIdx.x >> 5;

  float s = 0.f, ss = 0.f;
  for (int i = threadIdx.x; i < M; i += blockDim.x) {
    float v = x[base + i];
    s += v; ss += v * v;
  }
  #pragma unroll
  for (int o = 16; o > 0; o >>= 1) {
    s  += __shfl_xor(s,  o, 32);
    ss += __shfl_xor(ss, o, 32);
  }
  __shared__ float rs[8], rss[8];
  if (lane == 0) { rs[wid] = s; rss[wid] = ss; }
  __syncthreads();
  if (threadIdx.x == 0) {
    float t = 0.f, t2 = 0.f;
    for (int i = 0; i < 8; ++i) { t += rs[i]; t2 += rss[i]; }
    rs[0] = t; rss[0] = t2;
  }
  __syncthreads();
  float mean = rs[0] * (1.f / M);
  float var  = rss[0] * (1.f / M) - mean * mean;
  float inv  = rsqrtf(var + 1e-5f);

  // Transposed write: adjacent threads cover adjacent channels -> contiguous
  // 16-half store bursts into hn[n][c].
  for (int j = threadIdx.x; j < M; j += blockDim.x) {
    int n  = j >> 4;
    int cl = j & 15;
    int c  = g * CPG + cl;
    float v = (x[base + (size_t)cl * N_ + n] - mean) * inv;
    hn[((size_t)b * N_ + n) * C_ + c] = (_Float16)(v * gamma[c] + beta[c]);
  }
}

// ---------------------------------------------------------------------------
// Kernel 2: QKV projections. gridDim.z selects {q,k,v}.
// out[co][n] = sum_c W[co][c] * hn[n][c] + bias[co]
// q,k -> [B][N][C] (q pre-scaled by C^-0.5);  v -> [B][C][N].
// ---------------------------------------------------------------------------
__global__ void qkv_kernel(const _Float16* __restrict__ hn,
                           const float* __restrict__ qw, const float* __restrict__ qb,
                           const float* __restrict__ kw, const float* __restrict__ kb,
                           const float* __restrict__ vw, const float* __restrict__ vb,
                           _Float16* __restrict__ qo, _Float16* __restrict__ ko,
                           _Float16* __restrict__ vo) {
  int lane = threadIdx.x & 31;
  int tile = blockIdx.x * 8 + (threadIdx.x >> 5);   // 4096: 32 co-tiles x 128 n128-tiles
  int co0 = (tile >> 7) << 4;
  int ng  = (tile & 127) << 7;                      // global n over B*N, step 128
  int b   = ng / N_;
  int n0  = ng % N_;
  int mode = blockIdx.z;
  const float* W    = (mode == 0) ? qw : ((mode == 1) ? kw : vw);
  const float* bias = (mode == 0) ? qb : ((mode == 1) ? kb : vb);
  const _Float16* hb = hn + (size_t)b * N_ * C_;    // [n][c]

  v8f acc[NJ] = ZACC8;
  for (int k0 = 0; k0 < C_; k0 += 32) {
    v16h a = frag_a_f32(W, C_, co0, k0, lane);
    v16h bf[NJ];
    #pragma unroll
    for (int j = 0; j < NJ; ++j)
      bf[j] = frag_b_nmajor(hb, C_, k0, n0 + j * 16, lane);
    #pragma unroll
    for (int j = 0; j < NJ; ++j)
      acc[j] = WMMA_F16(a, bf[j], acc[j]);
  }
  int rb = (lane >> 4) << 3;
  if (mode == 2) {
    #pragma unroll
    for (int j = 0; j < NJ; ++j) {
      int col = n0 + j * 16 + (lane & 15);
      #pragma unroll
      for (int i = 0; i < 8; ++i) {
        int m = co0 + rb + i;
        vo[((size_t)b * C_ + m) * N_ + col] = (_Float16)(acc[j][i] + bias[m]);
      }
    }
  } else {
    const float sc = (mode == 0) ? 0.044194173824159216f : 1.f;  // 512^-0.5 on q
    _Float16* dst = (mode == 0) ? qo : ko;
    #pragma unroll
    for (int j = 0; j < NJ; ++j) {
      int col = n0 + j * 16 + (lane & 15);
      #pragma unroll
      for (int i = 0; i < 8; ++i) {
        int m = co0 + rb + i;
        dst[((size_t)b * N_ + col) * C_ + m] = (_Float16)((acc[j][i] + bias[m]) * sc);
      }
    }
  }
}

// ---------------------------------------------------------------------------
// Kernel 3: S = Qs @ K^T (per batch). Q,K f16 [N][C]; S f16 [N][N].
// 16(nq) x 128(nk) tile per wave.
// ---------------------------------------------------------------------------
__global__ void qk_kernel(const _Float16* __restrict__ q, const _Float16* __restrict__ k,
                          _Float16* __restrict__ S, int b) {
  int lane = threadIdx.x & 31;
  int tile = blockIdx.x * 8 + (threadIdx.x >> 5);   // 8192: 256 nq x 32 nk128
  int nq0 = (tile >> 5) << 4;
  int nk0 = (tile & 31) << 7;
  const _Float16* qb_ = q + (size_t)b * N_ * C_;    // [nq][c]
  const _Float16* kb_ = k + (size_t)b * N_ * C_;    // [nk][c] (B-operand, n-major)

  v8f acc[NJ] = ZACC8;
  for (int k0 = 0; k0 < C_; k0 += 32) {
    if (k0 + 32 < C_) {   // gfx1250 global_prefetch_b8 for next K-step
      __builtin_prefetch(qb_ + (size_t)(nq0 + (lane & 15)) * C_ + k0 + 32, 0, 0);
      __builtin_prefetch(kb_ + (size_t)(nk0 + (lane & 15)) * C_ + k0 + 32, 0, 0);
    }
    v16h a = frag_a_f16(qb_, C_, nq0, k0, lane);
    v16h bf[NJ];
    #pragma unroll
    for (int j = 0; j < NJ; ++j)
      bf[j] = frag_b_nmajor(kb_, C_, k0, nk0 + j * 16, lane);
    #pragma unroll
    for (int j = 0; j < NJ; ++j)
      acc[j] = WMMA_F16(a, bf[j], acc[j]);
  }
  int rb = (lane >> 4) << 3;
  #pragma unroll
  for (int j = 0; j < NJ; ++j) {
    int col = nk0 + j * 16 + (lane & 15);
    #pragma unroll
    for (int i = 0; i < 8; ++i)
      S[(size_t)(nq0 + rb + i) * N_ + col] = (_Float16)acc[j][i];
  }
}

// ---------------------------------------------------------------------------
// Kernel 4: row softmax over S (in place, f32 math). One block per row.
// ---------------------------------------------------------------------------
__global__ void softmax_kernel(_Float16* __restrict__ S) {
  _Float16* p = S + (size_t)blockIdx.x * N_;
  int lane = threadIdx.x & 31, wid = threadIdx.x >> 5;
  float v[16];
  float mx = -3.4e38f;
  #pragma unroll
  for (int j = 0; j < 16; ++j) {
    v[j] = (float)p[j * 256 + threadIdx.x];
    mx = fmaxf(mx, v[j]);
  }
  __shared__ float red[8];
  #pragma unroll
  for (int o = 16; o > 0; o >>= 1) mx = fmaxf(mx, __shfl_xor(mx, o, 32));
  if (lane == 0) red[wid] = mx;
  __syncthreads();
  if (threadIdx.x == 0) {
    float t = red[0];
    for (int i = 1; i < 8; ++i) t = fmaxf(t, red[i]);
    red[0] = t;
  }
  __syncthreads();
  mx = red[0];

  float sum = 0.f;
  #pragma unroll
  for (int j = 0; j < 16; ++j) { v[j] = __expf(v[j] - mx); sum += v[j]; }
  #pragma unroll
  for (int o = 16; o > 0; o >>= 1) sum += __shfl_xor(sum, o, 32);
  __syncthreads();
  if (lane == 0) red[wid] = sum;
  __syncthreads();
  if (threadIdx.x == 0) {
    float t = 0.f;
    for (int i = 0; i < 8; ++i) t += red[i];
    red[0] = t;
  }
  __syncthreads();
  float inv = 1.f / red[0];
  #pragma unroll
  for (int j = 0; j < 16; ++j)
    p[j * 256 + threadIdx.x] = (_Float16)(v[j] * inv);
}

// ---------------------------------------------------------------------------
// Kernel 5: O = P @ V (per batch). P f16 [N][N]; V f16 [C][N]; O f16 [B][N][C].
// 16(nq) x 128(c) tile per wave; K loop over all 4096 keys.
// ---------------------------------------------------------------------------
__global__ void pv_kernel(const _Float16* __restrict__ P, const _Float16* __restrict__ v,
                          _Float16* __restrict__ O, int b) {
  int lane = threadIdx.x & 31;
  int tile = blockIdx.x * 8 + (threadIdx.x >> 5);   // 1024: 256 nq x 4 c128
  int nq0 = (tile >> 2) << 4;
  int c0  = (tile & 3) << 7;
  const _Float16* vb_ = v + (size_t)b * C_ * N_;    // [c][nk] (B-operand, n-major in c)

  v8f acc[NJ] = ZACC8;
  for (int k0 = 0; k0 < N_; k0 += 32) {
    if (k0 + 32 < N_) {
      __builtin_prefetch(P + (size_t)(nq0 + (lane & 15)) * N_ + k0 + 32, 0, 0);
      __builtin_prefetch(vb_ + (size_t)(c0 + (lane & 15)) * N_ + k0 + 32, 0, 0);
    }
    v16h a = frag_a_f16(P, N_, nq0, k0, lane);      // P[nq][nk]
    v16h bf[NJ];
    #pragma unroll
    for (int j = 0; j < NJ; ++j)
      bf[j] = frag_b_nmajor(vb_, N_, k0, c0 + j * 16, lane);  // (nk, c) = V[c][nk]
    #pragma unroll
    for (int j = 0; j < NJ; ++j)
      acc[j] = WMMA_F16(a, bf[j], acc[j]);
  }
  int rb = (lane >> 4) << 3;
  #pragma unroll
  for (int j = 0; j < NJ; ++j) {
    int col = c0 + j * 16 + (lane & 15);
    #pragma unroll
    for (int i = 0; i < 8; ++i)
      O[((size_t)b * N_ + (nq0 + rb + i)) * C_ + col] = (_Float16)acc[j][i];
  }
}

// ---------------------------------------------------------------------------
// Kernel 6: final = x + proj_w @ O + proj_b.  O f16 [B][N][C]; out f32 [B][C][N].
// ---------------------------------------------------------------------------
__global__ void proj_kernel(const _Float16* __restrict__ O, const float* __restrict__ pw,
                            const float* __restrict__ pb, const float* __restrict__ x,
                            float* __restrict__ out) {
  int lane = threadIdx.x & 31;
  int tile = blockIdx.x * 8 + (threadIdx.x >> 5);   // 4096: 32 co x 128 n128
  int co0 = (tile >> 7) << 4;
  int ng  = (tile & 127) << 7;
  int b   = ng / N_;
  int n0  = ng % N_;
  const _Float16* Ob = O + (size_t)b * N_ * C_;     // [n][c]

  v8f acc[NJ] = ZACC8;
  for (int k0 = 0; k0 < C_; k0 += 32) {
    v16h a = frag_a_f32(pw, C_, co0, k0, lane);     // pw[co][c]
    v16h bf[NJ];
    #pragma unroll
    for (int j = 0; j < NJ; ++j)
      bf[j] = frag_b_nmajor(Ob, C_, k0, n0 + j * 16, lane);   // (c, n) = O[n][c]
    #pragma unroll
    for (int j = 0; j < NJ; ++j)
      acc[j] = WMMA_F16(a, bf[j], acc[j]);
  }
  int rb = (lane >> 4) << 3;
  #pragma unroll
  for (int j = 0; j < NJ; ++j) {
    int col = n0 + j * 16 + (lane & 15);
    #pragma unroll
    for (int i = 0; i < 8; ++i) {
      int m = co0 + rb + i;
      size_t idx = ((size_t)b * C_ + m) * N_ + col;
      out[idx] = x[idx] + acc[j][i] + pb[m];
    }
  }
}

// ---------------------------------------------------------------------------
// Host launcher. Workspace layout (halfs):
//   hn [B][N][C] | q [B][N][C] | k [B][N][C] | v [B][C][N] | O [B][N][C]
//   S  [N][N] (32 MB, reused per batch; stays in 192MB L2)
// ---------------------------------------------------------------------------
extern "C" void kernel_launch(void* const* d_in, const int* in_sizes, int n_in,
                              void* d_out, int out_size, void* d_ws, size_t ws_size,
                              hipStream_t stream) {
  (void)in_sizes; (void)n_in; (void)out_size; (void)ws_size;
  const float* x  = (const float*)d_in[0];
  const float* gg = (const float*)d_in[1];
  const float* gb = (const float*)d_in[2];
  const float* qw = (const float*)d_in[3];
  const float* qb = (const float*)d_in[4];
  const float* kw = (const float*)d_in[5];
  const float* kb = (const float*)d_in[6];
  const float* vw = (const float*)d_in[7];
  const float* vb = (const float*)d_in[8];
  const float* pw = (const float*)d_in[9];
  const float* pb = (const float*)d_in[10];
  float* out = (float*)d_out;

  const size_t T = (size_t)B_ * C_ * N_;
  _Float16* hn = (_Float16*)d_ws;
  _Float16* qh = hn + T;
  _Float16* kh = qh + T;
  _Float16* vh = kh + T;
  _Float16* Oh = vh + T;
  _Float16* Sh = Oh + T;

  gn_kernel<<<B_ * G_, 256, 0, stream>>>(x, gg, gb, hn);
  qkv_kernel<<<dim3(512, 1, 3), 256, 0, stream>>>(hn, qw, qb, kw, kb, vw, vb,
                                                  qh, kh, vh);
  for (int b = 0; b < B_; ++b) {
    qk_kernel<<<1024, 256, 0, stream>>>(qh, kh, Sh, b);
    softmax_kernel<<<N_, 256, 0, stream>>>(Sh);
    pv_kernel<<<128, 256, 0, stream>>>(Sh, vh, Oh, b);
  }
  proj_kernel<<<512, 256, 0, stream>>>(Oh, pw, pb, x, out);
}
